// Block_5360119185758
// MI455X (gfx1250) — matrix operand order repaired and checked by reading, not compiled, not run on hardware
//
#include <hip/hip_runtime.h>
#include <cstdint>
#include <cstddef>

typedef unsigned short u16;
typedef __attribute__((ext_vector_type(16))) __bf16 v16bf;
typedef __attribute__((ext_vector_type(8)))  float  v8f;
typedef int v4i_g __attribute__((vector_size(16)));   // GCC-style, matches builtin
typedef unsigned int u32x4 __attribute__((vector_size(16)));
typedef int i32x8 __attribute__((vector_size(32)));
typedef int i32x4 __attribute__((vector_size(16)));

#if defined(__has_builtin)
#if __has_builtin(__builtin_amdgcn_global_load_async_to_lds_b128) && \
    __has_builtin(__builtin_amdgcn_s_wait_asynccnt)
#define USE_ASYNC_LDS 1
#endif
#if __has_builtin(__builtin_amdgcn_permlane16)
#define USE_PERMLANE16 1
#endif
#if __has_builtin(__builtin_amdgcn_tensor_load_to_lds) && \
    __has_builtin(__builtin_amdgcn_s_wait_tensorcnt)
#define USE_TDM 1
#endif
#endif

union Frag {
  v16bf v;
  uint4 q[2];
};

__device__ __forceinline__ u16 f2bf(float f) {
  unsigned u = __float_as_uint(f);
  u += 0x7FFFu + ((u >> 16) & 1u);   // round-to-nearest-even
  return (u16)(u >> 16);
}

__device__ __forceinline__ v8f wmma_bf16(Frag a, Frag b, v8f c) {
  return __builtin_amdgcn_wmma_f32_16x16x32_bf16(
      /*neg_a=*/false, a.v, /*neg_b=*/false, b.v,
      /*c_mod=*/(short)0, c, /*reuse_a=*/false, /*reuse_b=*/false);
}

// 16B global->LDS copy: async (ASYNCcnt-tracked, bypasses VGPRs) when available
__device__ __forceinline__ void cp16_g2l(const u16* gsrc, u16* ldst) {
#ifdef USE_ASYNC_LDS
  __builtin_amdgcn_global_load_async_to_lds_b128(
      (__attribute__((address_space(1))) v4i_g*)(uintptr_t)gsrc,
      (__attribute__((address_space(3))) v4i_g*)(unsigned)(uintptr_t)ldst,
      0, 0);
#else
  *(uint4*)ldst = *(const uint4*)gsrc;
#endif
}
__device__ __forceinline__ void cp_wait() {
#ifdef USE_ASYNC_LDS
  __builtin_amdgcn_s_wait_asynccnt(0);
#endif
}

#ifdef USE_TDM
// Tensor Data Mover: DMA a 2D tile [rows][64 bf16] (row stride = stride_elems)
// from global into LDS at lds_off, padding each 128B row with 16B so the LDS
// row stride is 72 elements (conflict-free for the WMMA fragment loads).
// D# per CDNA5 ISA §8: group0 = control/lds/global addr, group1 = dims/tile.
// This toolchain exposes the 6-arg builtin: (g0, g1, g2, g3, g4, cpol).
__device__ __forceinline__ void tdm_load_2d(const u16* gbase, unsigned lds_off,
                                            int rows, int stride_elems) {
  const unsigned long long ga = (unsigned long long)(uintptr_t)gbase;
  u32x4 g0;
  g0[0] = 1u;                                          // count=1, user mode
  g0[1] = lds_off;                                     // lds_addr (bytes)
  g0[2] = (unsigned)(ga & 0xFFFFFFFFu);                // global_addr[31:0]
  g0[3] = (unsigned)((ga >> 32) & 0x01FFFFFFu)         // global_addr[56:32]
        | (2u << 30);                                  // type=2 ("image")
  i32x8 g1;
  g1[0] = (1 << 16)       // data_size = 2 bytes
        | (1 << 20)       // pad_enable
        | (4 << 22)       // pad_interval: pad after 8*2^4 = 128 bytes
        | (3 << 25);      // pad_amount: 4 DWORDs = 16 bytes
  g1[1] = (64 & 0xFFFF) << 16;                         // tensor_dim0[15:0]
  g1[2] = 0 | ((rows & 0xFFFF) << 16);                 // dim0[31:16] | dim1[15:0]
  g1[3] = 0 | (64 << 16);                              // dim1[31:16] | tile_dim0
  g1[4] = (rows & 0xFFFF);                             // tile_dim1 | tile_dim2=0
  g1[5] = stride_elems;                                // tensor_dim0_stride[31:0]
  g1[6] = 0;                                           // stride[47:32] | d1stride
  g1[7] = 0;
  const i32x4 z4 = {0, 0, 0, 0};                       // groups 2/3: 2D tensor
  const i32x8 z8 = {0, 0, 0, 0, 0, 0, 0, 0};
  __builtin_amdgcn_tensor_load_to_lds(g0, g1, z4, z4, z8, 0);
}
#endif

// XOR-butterfly within a 16-lane row, VALU-only via v_permlane16_b32
__device__ __forceinline__ float rowperm(float x, int s0, int s1) {
#ifdef USE_PERMLANE16
  int i = __float_as_int(x);
  i = __builtin_amdgcn_permlane16(i, i, s0, s1, false, false);
  return __int_as_float(i);
#else
  (void)s0; (void)s1;
  return x;
#endif
}
__device__ __forceinline__ float redmax16(float v) {
#ifdef USE_PERMLANE16
  v = fmaxf(v, rowperm(v, 0x67452301, 0xEFCDAB89));  // xor 1
  v = fmaxf(v, rowperm(v, 0x54761032, 0xDCFE98BA));  // xor 2
  v = fmaxf(v, rowperm(v, 0x32107654, 0xBA98FEDC));  // xor 4
  v = fmaxf(v, rowperm(v, 0xFEDCBA98, 0x76543210));  // xor 8
#else
  v = fmaxf(v, __shfl_xor(v, 1, 32));
  v = fmaxf(v, __shfl_xor(v, 2, 32));
  v = fmaxf(v, __shfl_xor(v, 4, 32));
  v = fmaxf(v, __shfl_xor(v, 8, 32));
#endif
  return v;
}
__device__ __forceinline__ float redsum16(float v) {
#ifdef USE_PERMLANE16
  v += rowperm(v, 0x67452301, 0xEFCDAB89);
  v += rowperm(v, 0x54761032, 0xDCFE98BA);
  v += rowperm(v, 0x32107654, 0xBA98FEDC);
  v += rowperm(v, 0xFEDCBA98, 0x76543210);
#else
  v += __shfl_xor(v, 1, 32);
  v += __shfl_xor(v, 2, 32);
  v += __shfl_xor(v, 4, 32);
  v += __shfl_xor(v, 8, 32);
#endif
  return v;
}

// ---------------------------------------------------------------------------
// Weight prep kernels
// ---------------------------------------------------------------------------
__global__ void build_qkv_w(const float* __restrict__ Wq, const float* __restrict__ Wk,
                            const float* __restrict__ Wv, u16* __restrict__ Wt) {
  int idx = blockIdx.x * 256 + threadIdx.x;     // idx = n*512 + c
  int n = idx >> 9, c = idx & 511;
  int part = n >> 9, hn = n & 511;
  int h = hn >> 6, d = hn & 63;
  const float* W = (part == 0) ? Wq : (part == 1) ? Wk : Wv;
  Wt[idx] = f2bf(W[((size_t)h * 512 + c) * 64 + d]);
}

__global__ void build_qkv_b(const float* __restrict__ bq, const float* __restrict__ bk,
                            const float* __restrict__ bv, float* __restrict__ bias) {
  int i = blockIdx.x * 256 + threadIdx.x;       // i < 1536
  int part = i >> 9, hn = i & 511;
  bias[i] = (part == 0) ? bq[hn] : (part == 1) ? bk[hn] : bv[hn];
}

__global__ void transpose_cvt(const float* __restrict__ W, u16* __restrict__ Wt,
                              int K, int N) {
  int idx = blockIdx.x * 256 + threadIdx.x;     // idx = n*K + k, grid exact
  int n = idx / K, k = idx % K;
  Wt[idx] = f2bf(W[(size_t)k * N + n]);
}

// ---------------------------------------------------------------------------
// LayerNorm over C=512, one row per 256-thread block, fp32 in -> bf16 out
// ---------------------------------------------------------------------------
__global__ __launch_bounds__(256) void layernorm512(
    const float* __restrict__ X, const float* __restrict__ g,
    const float* __restrict__ bta, u16* __restrict__ Out) {
  __shared__ float s1[256];
  __shared__ float s2[256];
  const int row = blockIdx.x, t = threadIdx.x;
  const float* xp = X + (size_t)row * 512;
  float a = xp[t], c = xp[t + 256];
  s1[t] = a + c;
  s2[t] = a * a + c * c;
  __syncthreads();
  for (int off = 128; off > 0; off >>= 1) {
    if (t < off) { s1[t] += s1[t + off]; s2[t] += s2[t + off]; }
    __syncthreads();
  }
  float mu  = s1[0] * (1.0f / 512.0f);
  float var = s2[0] * (1.0f / 512.0f) - mu * mu;
  float rs  = rsqrtf(var + 1e-5f);
  Out[(size_t)row * 512 + t]       = f2bf((a - mu) * rs * g[t]       + bta[t]);
  Out[(size_t)row * 512 + t + 256] = f2bf((c - mu) * rs * g[t + 256] + bta[t + 256]);
}

// ---------------------------------------------------------------------------
// Tiled bf16 WMMA GEMM: Out[M,N] = A[M,K] * Bt[N,K]^T (+bias)(+relu)(+resid)
// Block tile 128x128, 8 waves (2x4), wave tile 64x32.
// K stepped by 64 through double-buffered LDS. Tiles are staged by the Tensor
// Data Mover (wave 0 issues two D# descriptors, completion via TENSORcnt);
// fallback is per-lane async global->LDS. Copy of tile t+1 overlaps compute.
// ---------------------------------------------------------------------------
template <bool BIAS, bool RELU, bool RESID, bool OUTBF>
__global__ __launch_bounds__(256) void gemm_bf16(
    const u16* __restrict__ A, const u16* __restrict__ Bt,
    const float* __restrict__ bias, const float* __restrict__ resid,
    void* __restrict__ Out, int M, int N, int K) {
  __shared__ __align__(16) u16 sA[2][128 * 72];   // 64 K + 8 pad per row
  __shared__ __align__(16) u16 sB[2][128 * 72];
  const int tid  = threadIdx.x;
  const int lane = tid & 31;
  const int wave = tid >> 5;
  const int wm = wave >> 2, wn = wave & 3;
  const int n0 = blockIdx.x * 128;
  const int m0 = blockIdx.y * 128;
  const int l15 = lane & 15;

  v8f acc[4][2];
#pragma unroll
  for (int mt = 0; mt < 4; ++mt)
#pragma unroll
    for (int nt = 0; nt < 2; ++nt)
#pragma unroll
      for (int i = 0; i < 8; ++i) acc[mt][nt][i] = 0.0f;

  const int r0 = tid >> 3;   // fallback staging: 128 rows x 8 chunks of 16B
  const int ch = tid & 7;
  (void)r0; (void)ch;

  auto stage = [&](int bufi, int kk) {
#ifdef USE_TDM
    if (wave == 0) {
      tdm_load_2d(A  + (size_t)m0 * K + kk,
                  (unsigned)(uintptr_t)&sA[bufi][0], 128, K);
      tdm_load_2d(Bt + (size_t)n0 * K + kk,
                  (unsigned)(uintptr_t)&sB[bufi][0], 128, K);
    }
#else
#pragma unroll
    for (int i = 0; i < 4; ++i) {
      const int row = r0 + i * 32;
      cp16_g2l(A  + (size_t)(m0 + row) * K + kk + ch * 8, &sA[bufi][row * 72 + ch * 8]);
      cp16_g2l(Bt + (size_t)(n0 + row) * K + kk + ch * 8, &sB[bufi][row * 72 + ch * 8]);
    }
#endif
  };
  auto stage_wait = [&]() {
#ifdef USE_TDM
    if (wave == 0) __builtin_amdgcn_s_wait_tensorcnt(0);
#else
    cp_wait();
#endif
  };

  stage(0, 0);
  int buf = 0;
  for (int kk = 0; kk < K; kk += 64) {
    stage_wait();           // staging for `buf` complete (TENSORcnt / ASYNCcnt)
    __syncthreads();        // everyone's copies visible; prev readers done
    if (kk + 64 < K) stage(buf ^ 1, kk + 64);   // overlap next copy w/ compute
#pragma unroll
    for (int kf = 0; kf < 2; ++kf) {
      Frag af[4], bfr[2];
      const int c0 = kf * 32 + ((lane < 16) ? 0 : 8);   // A: K chunks c0, c0+16
#pragma unroll
      for (int mt = 0; mt < 4; ++mt) {
        const u16* p = &sA[buf][(wm * 64 + mt * 16 + l15) * 72];
        af[mt].q[0] = *(const uint4*)(p + c0);
        af[mt].q[1] = *(const uint4*)(p + c0 + 16);
      }
      const int b0 = kf * 32 + ((lane < 16) ? 0 : 8);   // B: K chunks b0, b0+16
#pragma unroll
      for (int nt = 0; nt < 2; ++nt) {
        const u16* p = &sB[buf][(wn * 32 + nt * 16 + l15) * 72 + b0];
        bfr[nt].q[0] = *(const uint4*)(p);
        bfr[nt].q[1] = *(const uint4*)(p + 16);
      }
#pragma unroll
      for (int mt = 0; mt < 4; ++mt)
#pragma unroll
        for (int nt = 0; nt < 2; ++nt)
          acc[mt][nt] = wmma_bf16(af[mt], bfr[nt], acc[mt][nt]);
    }
    __syncthreads();        // done reading `buf`; next iter may overwrite it
    buf ^= 1;
  }

  // epilogue: C layout lane = col, VGPR j = row j (lanes<16) / j+8 (lanes>=16)
#pragma unroll
  for (int nt = 0; nt < 2; ++nt) {
    const int n = n0 + wn * 32 + nt * 16 + l15;
    const float bvv = BIAS ? bias[n] : 0.0f;
#pragma unroll
    for (int mt = 0; mt < 4; ++mt) {
#pragma unroll
      for (int j = 0; j < 8; ++j) {
        const int m = m0 + wm * 64 + mt * 16 + ((lane < 16) ? j : j + 8);
        float v = acc[mt][nt][j] + bvv;
        if (RELU)  v = fmaxf(v, 0.0f);
        if (RESID) v += resid[(size_t)m * N + n];
        if (OUTBF) ((u16*)Out)[(size_t)m * N + n]  = f2bf(v);
        else       ((float*)Out)[(size_t)m * N + n] = v;
      }
    }
  }
}

// ---------------------------------------------------------------------------
// Flash attention, causal. QKV bf16 [B*T, 1536] (cols: Q|K|V each h*64+d).
// Grid (B*H, T/128). 8 waves x 16 Q-rows. K tile async-staged, V transposed
// on store. Online softmax with VALU-only 16-lane butterflies.
// ---------------------------------------------------------------------------
__global__ __launch_bounds__(256) void attn_flash(
    const u16* __restrict__ QKV, u16* __restrict__ Og) {
  constexpr int T = 2048, LDQ = 1536;
  __shared__ __align__(16) u16 Kl[64 * 72];       // K tile [kv][d]
  __shared__ __align__(16) u16 Vt[64 * 72];       // V tile transposed [d][kv]
  __shared__ __align__(16) u16 Pl[8 * 16 * 72];   // per-wave P staging

  const int tid = threadIdx.x, lane = tid & 31, wave = tid >> 5;
  const int bh = blockIdx.x;
  const int b = bh >> 3, hh = bh & 7;
  const int qb = blockIdx.y;
  const int l15 = lane & 15;
  const int qbase = qb * 128 + wave * 16;

  // Q fragments (16x64 -> two 16x32 A-frags), loaded straight from global
  Frag qf[2];
  {
    const int r = b * T + qbase + l15;
    const u16* base = QKV + (size_t)r * LDQ + hh * 64;
    const int c0 = (lane < 16) ? 0 : 8;
#pragma unroll
    for (int kf = 0; kf < 2; ++kf) {
      qf[kf].q[0] = *(const uint4*)(base + kf * 32 + c0);
      qf[kf].q[1] = *(const uint4*)(base + kf * 32 + c0 + 16);
    }
  }

  v8f o[4];
#pragma unroll
  for (int nt = 0; nt < 4; ++nt)
#pragma unroll
    for (int i = 0; i < 8; ++i) o[nt][i] = 0.0f;
  float m_i[8], l_i[8];
#pragma unroll
  for (int j = 0; j < 8; ++j) { m_i[j] = -1e30f; l_i[j] = 0.0f; }

  const int nblk = 2 * (qb + 1);                  // causal KV extent
  const int r0 = tid >> 3;                        // staging: 64 rows x 8 chunks
  const int sch = tid & 7;
  u16* pw = Pl + wave * 16 * 72;

  for (int jb = 0; jb < nblk; ++jb) {
    const int kv0 = jb * 64;
    __syncthreads();
#pragma unroll
    for (int i = 0; i < 2; ++i) {
      const int row = r0 + i * 32;
      const size_t gr = (size_t)(b * T + kv0 + row) * LDQ + hh * 64 + sch * 8;
      cp16_g2l(QKV + gr + 512, Kl + row * 72 + sch * 8);   // K: async path
      union { uint4 q; u16 s[8]; } vv;                     // V: transpose-store
      vv.q = *(const uint4*)(QKV + gr + 1024);
#pragma unroll
      for (int e = 0; e < 8; ++e)
        Vt[(sch * 8 + e) * 72 + row] = vv.s[e];
    }
    cp_wait();
    __syncthreads();

    // S = Q * K^T  (4 col tiles x 2 K-steps)
    v8f s[4];
#pragma unroll
    for (int nt = 0; nt < 4; ++nt) {
#pragma unroll
      for (int i = 0; i < 8; ++i) s[nt][i] = 0.0f;
      const int b0 = (lane < 16) ? 0 : 8;
#pragma unroll
      for (int kf = 0; kf < 2; ++kf) {
        Frag kb;
        const u16* p = Kl + (nt * 16 + l15) * 72 + kf * 32 + b0;
        kb.q[0] = *(const uint4*)(p);
        kb.q[1] = *(const uint4*)(p + 16);
        s[nt] = wmma_bf16(qf[kf], kb, s[nt]);
      }
      // scale 1/sqrt(64) + causal mask
      const int kvi = kv0 + nt * 16 + l15;
#pragma unroll
      for (int j = 0; j < 8; ++j) {
        const int qi = qbase + ((lane < 16) ? j : j + 8);
        s[nt][j] = (kvi > qi) ? -1e30f : s[nt][j] * 0.125f;
      }
    }

    // online softmax per row (VALU butterflies across each 16-lane half)
#pragma unroll
    for (int j = 0; j < 8; ++j) {
      float rm = fmaxf(fmaxf(s[0][j], s[1][j]), fmaxf(s[2][j], s[3][j]));
      rm = redmax16(rm);
      const float mn = fmaxf(m_i[j], rm);
      const float alpha = __expf(m_i[j] - mn);
      float rs = 0.0f;
#pragma unroll
      for (int nt = 0; nt < 4; ++nt) {
        float p = __expf(s[nt][j] - mn);
        s[nt][j] = p;
        rs += p;
      }
      rs = redsum16(rs);
      l_i[j] = l_i[j] * alpha + rs;
      m_i[j] = mn;
#pragma unroll
      for (int nt = 0; nt < 4; ++nt) o[nt][j] *= alpha;
    }

    // P (C layout) -> LDS -> A-fragment layout (wave-private region)
#pragma unroll
    for (int nt = 0; nt < 4; ++nt)
#pragma unroll
      for (int j = 0; j < 8; ++j) {
        const int rloc = (lane < 16) ? j : j + 8;
        pw[rloc * 72 + nt * 16 + l15] = f2bf(s[nt][j]);
      }
    Frag pf[2];
    {
      const u16* pr = pw + l15 * 72;
      const int c0 = (lane < 16) ? 0 : 8;
#pragma unroll
      for (int kf = 0; kf < 2; ++kf) {
        pf[kf].q[0] = *(const uint4*)(pr + kf * 32 + c0);
        pf[kf].q[1] = *(const uint4*)(pr + kf * 32 + c0 + 16);
      }
    }

    // O += P * V
    const int b0 = (lane < 16) ? 0 : 8;
#pragma unroll
    for (int nt = 0; nt < 4; ++nt)
#pragma unroll
      for (int kf = 0; kf < 2; ++kf) {
        Frag vb;
        const u16* p = Vt + (nt * 16 + l15) * 72 + kf * 32 + b0;
        vb.q[0] = *(const uint4*)(p);
        vb.q[1] = *(const uint4*)(p + 16);
        o[nt] = wmma_bf16(pf[kf], vb, o[nt]);
      }
  }

  // normalize and store (bf16, head-concat layout == [B*T, 512])
#pragma unroll
  for (int j = 0; j < 8; ++j) {
    const float inv = 1.0f / l_i[j];
    const int r = b * T + qbase + ((lane < 16) ? j : j + 8);
#pragma unroll
    for (int nt = 0; nt < 4; ++nt)
      Og[(size_t)r * 512 + hh * 64 + nt * 16 + l15] = f2bf(o[nt][j] * inv);
  }
}

// ---------------------------------------------------------------------------
// Host orchestration
// ---------------------------------------------------------------------------
extern "C" void kernel_launch(void* const* d_in, const int* in_sizes, int n_in,
                              void* d_out, int out_size, void* d_ws, size_t ws_size,
                              hipStream_t stream) {
  (void)in_sizes; (void)n_in; (void)out_size; (void)ws_size;
  const float* x    = (const float*)d_in[0];
  const float* ln1g = (const float*)d_in[1];
  const float* ln1b = (const float*)d_in[2];
  const float* Wq   = (const float*)d_in[3];
  const float* bq   = (const float*)d_in[4];
  const float* Wk   = (const float*)d_in[5];
  const float* bk   = (const float*)d_in[6];
  const float* Wv   = (const float*)d_in[7];
  const float* bv   = (const float*)d_in[8];
  const float* Wp   = (const float*)d_in[9];
  const float* bp   = (const float*)d_in[10];
  const float* ln2g = (const float*)d_in[11];
  const float* ln2b = (const float*)d_in[12];
  const float* W1   = (const float*)d_in[13];
  const float* b1   = (const float*)d_in[14];
  const float* W2   = (const float*)d_in[15];
  const float* b2   = (const float*)d_in[16];

  const int M = 8192;   // B*T

  char* ws = (char*)d_ws;
  auto alloc = [&](size_t bytes) {
    char* p = ws;
    ws += (bytes + 255) & ~(size_t)255;
    return p;
  };
  u16*   hb     = (u16*)  alloc((size_t)M * 512 * 2);
  u16*   qkv    = (u16*)  alloc((size_t)M * 1536 * 2);
  u16*   attno  = (u16*)  alloc((size_t)M * 512 * 2);
  float* x2     = (float*)alloc((size_t)M * 512 * 4);
  u16*   h2     = (u16*)  alloc((size_t)M * 512 * 2);
  u16*   ff1    = (u16*)  alloc((size_t)M * 2048 * 2);
  u16*   wqkv_t = (u16*)  alloc((size_t)1536 * 512 * 2);
  float* bqkv   = (float*)alloc((size_t)1536 * 4);
  u16*   wp_t   = (u16*)  alloc((size_t)512 * 512 * 2);
  u16*   w1_t   = (u16*)  alloc((size_t)2048 * 512 * 2);
  u16*   w2_t   = (u16*)  alloc((size_t)512 * 2048 * 2);

  // weight prep
  build_qkv_w<<<(1536 * 512) / 256, 256, 0, stream>>>(Wq, Wk, Wv, wqkv_t);
  build_qkv_b<<<1536 / 256, 256, 0, stream>>>(bq, bk, bv, bqkv);
  transpose_cvt<<<(512 * 512) / 256, 256, 0, stream>>>(Wp, wp_t, 512, 512);
  transpose_cvt<<<(2048 * 512) / 256, 256, 0, stream>>>(W1, w1_t, 512, 2048);
  transpose_cvt<<<(512 * 2048) / 256, 256, 0, stream>>>(W2, w2_t, 2048, 512);

  // h = LN1(x)
  layernorm512<<<M, 256, 0, stream>>>(x, ln1g, ln1b, hb);

  // QKV = h @ Wqkv + bqkv   (bf16 out)
  gemm_bf16<true, false, false, true>
      <<<dim3(1536 / 128, M / 128), 256, 0, stream>>>(
          hb, wqkv_t, bqkv, nullptr, qkv, M, 1536, 512);

  // attention (flash, causal)
  attn_flash<<<dim3(32, 16), 256, 0, stream>>>(qkv, attno);

  // x2 = x + attno @ Wp + bp   (fp32 out)
  gemm_bf16<true, false, true, false>
      <<<dim3(512 / 128, M / 128), 256, 0, stream>>>(
          attno, wp_t, bp, x, x2, M, 512, 512);

  // h2 = LN2(x2)
  layernorm512<<<M, 256, 0, stream>>>(x2, ln2g, ln2b, h2);

  // ff1 = relu(h2 @ W1 + b1)   (bf16 out)
  gemm_bf16<true, true, false, true>
      <<<dim3(2048 / 128, M / 128), 256, 0, stream>>>(
          h2, w1_t, b1, nullptr, ff1, M, 2048, 512);

  // out = x2 + ff1 @ W2 + b2   (fp32 out)
  gemm_bf16<true, false, true, false>
      <<<dim3(512 / 128, M / 128), 256, 0, stream>>>(
          ff1, w2_t, b2, x2, (float*)d_out, M, 512, 2048);
}
